// MaskedAttention_4595615007196
// MI455X (gfx1250) — compile-verified, matrix operand order checked by reading, not run on hardware
//
#include <hip/hip_runtime.h>

#define DEV __device__ __forceinline__

typedef __attribute__((ext_vector_type(16))) __bf16 v16bf;
typedef __attribute__((ext_vector_type(8)))  __bf16 v8bf;
typedef __attribute__((ext_vector_type(8)))  float  v8f;
typedef int v4i_async __attribute__((vector_size(16)));

#define NEG_HUGE (-3.402823466e38f)

// ---- constants for this problem ----
#define BB   2
#define HH   6
#define NN   4096
#define CC   384
#define HD   64
#define C3   (3*CC)      // 1152
#define ROWS (BB*NN)     // 8192

#if defined(__has_builtin)
#if __has_builtin(__builtin_amdgcn_global_load_async_to_lds_b128) && \
    __has_builtin(__builtin_amdgcn_s_wait_asynccnt)
#define HAVE_ASYNC 1
#endif
#if __has_builtin(__builtin_amdgcn_tensor_load_to_lds) && \
    __has_builtin(__builtin_amdgcn_s_wait_tensorcnt)
#define HAVE_TDM 1
#endif
#if __has_builtin(__builtin_amdgcn_sched_barrier)
#define SCHED_FENCE() __builtin_amdgcn_sched_barrier(0)
#endif
#endif
#ifndef SCHED_FENCE
#define SCHED_FENCE()
#endif

// Async ops outstanding per wave per K/V prefetch (for the s_wait ledger)
#ifdef HAVE_TDM
#define ASYNC_PER_PF 4   // K tile only (V via TDM)
#else
#define ASYNC_PER_PF 8   // K + V tiles
#endif

union FragU { v16bf v; v8bf h[2]; };

// Load one 16-bit A/B WMMA fragment (v16bf) given a pointer to the K-base of
// this lane's row (A) or column (B). Layout per ISA 7.12.2 (16-bit A 16x32):
//   lane<16 : elems 0..7 = K[0..7],   elems 8..15 = K[16..23]
//   lane>=16: elems 0..7 = K[8..15],  elems 8..15 = K[24..31]
DEV v16bf load_frag(const __bf16* p, int hlf) {
  FragU f;
  f.h[0] = *(const v8bf*)(p + 8 * hlf);
  f.h[1] = *(const v8bf*)(p + 16 + 8 * hlf);
  return f.v;
}

DEV v8f wmma_bf16(v16bf a, v16bf b, v8f c) {
  return __builtin_amdgcn_wmma_f32_16x16x32_bf16(false, a, false, b, (short)0, c,
                                                 false, false);
}

// 16-byte global -> LDS copy; async (ASYNCcnt-tracked) when available.
DEV void cp16(const __bf16* g, __bf16* l) {
#ifdef HAVE_ASYNC
  __builtin_amdgcn_global_load_async_to_lds_b128(
      (__attribute__((address_space(1))) v4i_async*)(g),
      (__attribute__((address_space(3))) v4i_async*)(l), 0, 0);
#else
  *(v8bf*)l = *(const v8bf*)g;
#endif
}

// Stage one 64x64 bf16 tile (row stride src_row_stride elems) into LDS with
// 4 x b128 per thread (128 threads) => ASYNCcnt += 4 per wave.
DEV void stage_tile_cp(const __bf16* src, int src_row_stride, __bf16* dst, int tid) {
#pragma unroll
  for (int i = 0; i < 4; ++i) {
    const int c = tid + 128 * i;          // 512 16B-chunks per tile
    const int row = c >> 3;
    const int off = (c & 7) * 8;
    cp16(src + row * src_row_stride + off, dst + row * HD + off);
  }
}

#ifdef HAVE_TDM
typedef unsigned int tdm_v4u __attribute__((vector_size(16)));
typedef int          tdm_v8i __attribute__((vector_size(32)));
typedef int          tdm_v4i __attribute__((vector_size(16)));

// One-wave TDM load of a 2D bf16 tile (tile_w x tile_h, global row stride
// row_stride elems) into LDS. D# fields per CDNA5 ISA ch.8:
//   g0: count=1 | lds_addr | global_addr(56:0) | type=2
//   g1: data_size=1(2B), tensor_dim0=row_stride, tensor_dim1=tile_h,
//       tile_dim0=tile_w, tile_dim1=tile_h, tensor_dim0_stride=row_stride
// Toolchain builtin is the 6-arg form: (v4u, v8i, v4i, v4i, v8i, i32 cpol).
DEV void tdm_load_2d(const __bf16* gsrc, __bf16* ldst, unsigned tile_w,
                     unsigned tile_h, unsigned row_stride) {
  const unsigned long long ga = (unsigned long long)(const void*)gsrc;
  const unsigned la = (unsigned)(unsigned long long)
      (__attribute__((address_space(3))) void*)(void*)ldst;
  tdm_v4u g0;
  g0[0] = 1u;                                            // count=1, user mode
  g0[1] = la;                                            // lds_addr (bytes)
  g0[2] = (unsigned)(ga & 0xFFFFFFFFu);                  // global_addr lo
  g0[3] = (unsigned)((ga >> 32) & 0x01FFFFFFu) | (2u << 30);  // addr hi | type=2
  tdm_v8i g1;
  g1[0] = (int)(1u << 16);                               // data_size=1 (2B)
  g1[1] = (int)((row_stride & 0xFFFFu) << 16);           // tensor_dim0[15:0]
  g1[2] = (int)(((row_stride >> 16) & 0xFFFFu)           // tensor_dim0[31:16]
              | ((tile_h & 0xFFFFu) << 16));             // tensor_dim1[15:0]
  g1[3] = (int)(((tile_h >> 16) & 0xFFFFu)               // tensor_dim1[31:16]
              | ((tile_w & 0xFFFFu) << 16));             // tile_dim0
  g1[4] = (int)(tile_h & 0xFFFFu);                       // tile_dim1; tile_dim2=0
  g1[5] = (int)row_stride;                               // tensor_dim0_stride lo
  g1[6] = 0;                                             // stride hi; dim1_stride=0
  g1[7] = 0;
  const tdm_v4i z4 = {0, 0, 0, 0};
  const tdm_v8i z8 = {0, 0, 0, 0, 0, 0, 0, 0};
  __builtin_amdgcn_tensor_load_to_lds(g0, g1, z4, z4, z8, 0);
}
#endif

// ---------------- f32 -> bf16 convert ----------------
__global__ void cvt_bf16_k(const float* __restrict__ in, __bf16* __restrict__ out, int n) {
  int i = blockIdx.x * 256 + threadIdx.x;
  if (i < n) out[i] = (__bf16)in[i];
}

// ---------------- QKV GEMM: xb[8192x384] * wq[1152x384]^T ----------------
// grid (128, 18), block 128 (4 waves). Register-double-buffered fragments;
// sched_barrier pins the prefetch above the WMMAs (load/compute overlap).
__global__ __launch_bounds__(128)
void qkv_gemm_k(const __bf16* __restrict__ xb, const __bf16* __restrict__ wq,
                __bf16* __restrict__ qb, __bf16* __restrict__ kb,
                __bf16* __restrict__ vtb) {
  const int w    = threadIdx.x >> 5;
  const int lane = threadIdx.x & 31;
  const int hlf  = lane >> 4;
  const int ll   = lane & 15;
  const int m0   = blockIdx.x * 64 + w * 16;
  const int n0   = blockIdx.y * 64;
  const __bf16* Arow = xb + (m0 + ll) * CC;

  v8f acc[4] = {{}, {}, {}, {}};
  v16bf a[2], bm[2][4];
  a[0] = load_frag(Arow, hlf);
#pragma unroll
  for (int j = 0; j < 4; ++j)
    bm[0][j] = load_frag(wq + (n0 + 16 * j + ll) * CC, hlf);

  const int NIT = CC / 32;  // 12
#pragma unroll
  for (int it = 0; it < NIT; ++it) {
    const int cur = it & 1, nxt = cur ^ 1;
    if (it + 1 < NIT) {                      // prefetch next k-step
      const int kn = 32 * (it + 1);
      a[nxt] = load_frag(Arow + kn, hlf);
#pragma unroll
      for (int j = 0; j < 4; ++j)
        bm[nxt][j] = load_frag(wq + (n0 + 16 * j + ll) * CC + kn, hlf);
    }
    SCHED_FENCE();                           // keep prefetch above the WMMAs
#pragma unroll
    for (int j = 0; j < 4; ++j) acc[j] = wmma_bf16(a[cur], bm[cur][j], acc[j]);
    SCHED_FENCE();
  }

#pragma unroll
  for (int j = 0; j < 4; ++j) {
    const int c3 = n0 + 16 * j + ll;
    const int which = c3 / CC;         // 0=q 1=k 2=v (uniform within a j-tile)
    const int c = c3 - which * CC;
    const int h = c >> 6;
    const int d = c & 63;
#pragma unroll
    for (int r = 0; r < 8; ++r) {
      const int row = m0 + r + 8 * hlf;
      const int b = row >> 12;
      const int n = row & (NN - 1);
      const int bh = b * HH + h;
      const __bf16 val = (__bf16)acc[j][r];
      if (which == 0)      qb [(bh * NN + n) * HD + d] = val;
      else if (which == 1) kb [(bh * NN + n) * HD + d] = val;
      else                 vtb[(bh * HD + d) * NN + n] = val;   // V transposed
    }
  }
}

// ---------------- Flash attention ----------------
// grid (64 q-blocks, 12 b*h), block 128 (4 waves, 16 q-rows each).
// K tile: async global->LDS copies (all waves). V tile: TDM tensor_load_to_lds
// (wave 0 only) when available. Both double-buffered.
__global__ __launch_bounds__(128)
void attn_flash_k(const __bf16* __restrict__ qb, const __bf16* __restrict__ kb,
                  const __bf16* __restrict__ vtb,
                  const unsigned char* __restrict__ mask_all,
                  __bf16* __restrict__ ob) {
  __shared__ __align__(16) __bf16 sK[2][64 * HD];   // 2 x 8KB
  __shared__ __align__(16) __bf16 sV[2][64 * HD];   // 2 x 8KB (Vt tile: [dim][key])
  __shared__ __align__(16) __bf16 sP[4][16 * 64];   // per-wave P staging

  const int tid  = threadIdx.x;
  const int w    = tid >> 5;
  const int lane = tid & 31;
  const int hlf  = lane >> 4;
  const int ll   = lane & 15;
  const int bh   = blockIdx.y;
  const int b    = bh / HH;
  const int h    = bh - b * HH;
  const int q0   = blockIdx.x * 64 + w * 16;

  const __bf16* Q  = qb  + bh * NN * HD;
  const __bf16* Kp = kb  + bh * NN * HD;
  const __bf16* Vt = vtb + bh * HD * NN;
  const unsigned char* mb = mask_all + b * NN;

  // Q fragments (reused for all key blocks)
  v16bf qf0 = load_frag(Q + (q0 + ll) * HD + 0,  hlf);
  v16bf qf1 = load_frag(Q + (q0 + ll) * HD + 32, hlf);

  bool qm[8];
#pragma unroll
  for (int r = 0; r < 8; ++r) qm[r] = mb[q0 + r + 8 * hlf] != 0;

  float mrow[8], lrow[8];
#pragma unroll
  for (int r = 0; r < 8; ++r) { mrow[r] = NEG_HUGE; lrow[r] = 0.f; }
  v8f oacc[4] = {{}, {}, {}, {}};

  __bf16* myP = &sP[w][0];

  // prologue prefetch of block 0
  stage_tile_cp(Kp, HD, sK[0], tid);
#ifdef HAVE_TDM
  if (w == 0) tdm_load_2d(Vt, sV[0], 64, 64, NN);
#else
  stage_tile_cp(Vt, NN, sV[0], tid);
#endif

  const int nblk = NN / 64;
  for (int i = 0; i < nblk; ++i) {
    const int k0  = i * 64;
    const int cur = i & 1;
    if (i + 1 < nblk) {
      stage_tile_cp(Kp + (k0 + 64) * HD, HD, sK[cur ^ 1], tid);
#ifdef HAVE_TDM
      if (w == 0) tdm_load_2d(Vt + k0 + 64, sV[cur ^ 1], 64, 64, NN);
#else
      stage_tile_cp(Vt + k0 + 64, NN, sV[cur ^ 1], tid);
#endif
#ifdef HAVE_ASYNC
      __builtin_amdgcn_s_wait_asynccnt(ASYNC_PER_PF);  // retire previous prefetch
#endif
#ifdef HAVE_TDM
      if (w == 0) __builtin_amdgcn_s_wait_tensorcnt(1);
#endif
    } else {
#ifdef HAVE_ASYNC
      __builtin_amdgcn_s_wait_asynccnt(0);
#endif
#ifdef HAVE_TDM
      if (w == 0) __builtin_amdgcn_s_wait_tensorcnt(0);
#endif
    }
    __syncthreads();

    const __bf16* kT = &sK[cur][0];
    const __bf16* vT = &sV[cur][0];

    // ---- S = Q K^T (16 x 64 strip), K-dim = head dim (2 steps of 32) ----
    v8f s[4] = {{}, {}, {}, {}};
#pragma unroll
    for (int j = 0; j < 4; ++j) {
      const __bf16* kr = kT + (16 * j + ll) * HD;
      s[j] = wmma_bf16(qf0, load_frag(kr + 0,  hlf), s[j]);
      s[j] = wmma_bf16(qf1, load_frag(kr + 32, hlf), s[j]);
    }

    bool km[4];
#pragma unroll
    for (int j = 0; j < 4; ++j) km[j] = mb[k0 + 16 * j + ll] != 0;

    // scale + pair mask (reference uses -FLT_MAX, not -inf)
    float sc[4][8];
#pragma unroll
    for (int j = 0; j < 4; ++j)
#pragma unroll
      for (int r = 0; r < 8; ++r) {
        float v = s[j][r] * 0.125f;                 // SCALE = 64^-0.5
        sc[j][r] = (qm[r] && km[j]) ? v : NEG_HUGE;
      }

    // ---- online softmax (row stats in 16-lane halves: VGPR r <-> row r/r+8)
    float alpha[8];
#pragma unroll
    for (int r = 0; r < 8; ++r) {
      float mx = fmaxf(fmaxf(sc[0][r], sc[1][r]), fmaxf(sc[2][r], sc[3][r]));
#pragma unroll
      for (int d = 1; d < 16; d <<= 1) mx = fmaxf(mx, __shfl_xor(mx, d, 32));
      float mnew = fmaxf(mrow[r], mx);
      alpha[r] = __expf(mrow[r] - mnew);
      mrow[r] = mnew;
      float rs = 0.f;
#pragma unroll
      for (int j = 0; j < 4; ++j) {
        float p = __expf(sc[j][r] - mnew);
        sc[j][r] = p;
        rs += p;
      }
#pragma unroll
      for (int d = 1; d < 16; d <<= 1) rs += __shfl_xor(rs, d, 32);
      lrow[r] = lrow[r] * alpha[r] + rs;
    }

    // ---- stage P (C-layout -> row-major LDS) and rescale O ----
#pragma unroll
    for (int j = 0; j < 4; ++j)
#pragma unroll
      for (int r = 0; r < 8; ++r) {
        myP[(r + 8 * hlf) * 64 + 16 * j + ll] = (__bf16)sc[j][r];
        oacc[j][r] *= alpha[r];
      }

    // ---- O += P * V (K-dim = 64 keys, 2 steps of 32) ----
#pragma unroll
    for (int s2 = 0; s2 < 2; ++s2) {
      v16bf pf = load_frag(myP + ll * 64 + 32 * s2, hlf);   // A row = q row ll
#pragma unroll
      for (int j = 0; j < 4; ++j) {
        v16bf vf = load_frag(vT + (16 * j + ll) * HD + 32 * s2, hlf);
        oacc[j] = wmma_bf16(pf, vf, oacc[j]);
      }
    }
    __syncthreads();
  }

  // ---- normalize and write O strip as bf16 [row][h*64+d] ----
#pragma unroll
  for (int r = 0; r < 8; ++r) {
    const float inv = 1.0f / lrow[r];
    const int row = b * NN + q0 + r + 8 * hlf;
#pragma unroll
    for (int j = 0; j < 4; ++j) {
      const int col = h * HD + 16 * j + ll;
      ob[row * CC + col] = (__bf16)(oacc[j][r] * inv);
    }
  }
}

// ---------------- Output projection: ob[8192x384]*wp[384x384]^T + bias ----------------
// grid (128, 6), block 128.
__global__ __launch_bounds__(128)
void proj_gemm_k(const __bf16* __restrict__ ob, const __bf16* __restrict__ wp,
                 const float* __restrict__ bias, float* __restrict__ out) {
  const int w    = threadIdx.x >> 5;
  const int lane = threadIdx.x & 31;
  const int hlf  = lane >> 4;
  const int ll   = lane & 15;
  const int m0   = blockIdx.x * 64 + w * 16;
  const int n0   = blockIdx.y * 64;
  const __bf16* Arow = ob + (m0 + ll) * CC;

  v8f acc[4] = {{}, {}, {}, {}};
  v16bf a[2], bm[2][4];
  a[0] = load_frag(Arow, hlf);
#pragma unroll
  for (int j = 0; j < 4; ++j)
    bm[0][j] = load_frag(wp + (n0 + 16 * j + ll) * CC, hlf);

  const int NIT = CC / 32;
#pragma unroll
  for (int it = 0; it < NIT; ++it) {
    const int cur = it & 1, nxt = cur ^ 1;
    if (it + 1 < NIT) {
      const int kn = 32 * (it + 1);
      a[nxt] = load_frag(Arow + kn, hlf);
#pragma unroll
      for (int j = 0; j < 4; ++j)
        bm[nxt][j] = load_frag(wp + (n0 + 16 * j + ll) * CC + kn, hlf);
    }
    SCHED_FENCE();
#pragma unroll
    for (int j = 0; j < 4; ++j) acc[j] = wmma_bf16(a[cur], bm[cur][j], acc[j]);
    SCHED_FENCE();
  }

#pragma unroll
  for (int j = 0; j < 4; ++j) {
    const int col = n0 + 16 * j + ll;
    const float bv = bias[col];
#pragma unroll
    for (int r = 0; r < 8; ++r) {
      const int row = m0 + r + 8 * hlf;
      out[row * CC + col] = acc[j][r] + bv;
    }
  }
}

// ---------------- host launch ----------------
extern "C" void kernel_launch(void* const* d_in, const int* in_sizes, int n_in,
                              void* d_out, int out_size, void* d_ws, size_t ws_size,
                              hipStream_t stream) {
  const float*         x      = (const float*)d_in[0];
  const unsigned char* mask   = (const unsigned char*)d_in[1];  // bool [B,N]
  const float*         w_qkv  = (const float*)d_in[2];
  const float*         w_proj = (const float*)d_in[3];
  const float*         b_proj = (const float*)d_in[4];
  float*               out    = (float*)d_out;

  char* ws = (char*)d_ws;
  size_t off = 0;
  auto alloc = [&](size_t bytes) -> void* {
    void* p = ws + off;
    off += (bytes + 255) & ~(size_t)255;
    return p;
  };

  __bf16* xb  = (__bf16*)alloc((size_t)ROWS * CC * 2);
  __bf16* wqb = (__bf16*)alloc((size_t)C3 * CC * 2);
  __bf16* wpb = (__bf16*)alloc((size_t)CC * CC * 2);
  __bf16* qb  = (__bf16*)alloc((size_t)BB * HH * NN * HD * 2);
  __bf16* kb  = (__bf16*)alloc((size_t)BB * HH * NN * HD * 2);
  __bf16* vtb = (__bf16*)alloc((size_t)BB * HH * HD * NN * 2);
  __bf16* ob  = (__bf16*)alloc((size_t)ROWS * CC * 2);

  const int nx = ROWS * CC, nwq = C3 * CC, nwp = CC * CC;
  cvt_bf16_k<<<(nx  + 255) / 256, 256, 0, stream>>>(x,      xb,  nx);
  cvt_bf16_k<<<(nwq + 255) / 256, 256, 0, stream>>>(w_qkv,  wqb, nwq);
  cvt_bf16_k<<<(nwp + 255) / 256, 256, 0, stream>>>(w_proj, wpb, nwp);

  qkv_gemm_k<<<dim3(ROWS / 64, C3 / 64), 128, 0, stream>>>(xb, wqb, qb, kb, vtb);
  attn_flash_k<<<dim3(NN / 64, BB * HH), 128, 0, stream>>>(qb, kb, vtb, mask, ob);
  proj_gemm_k<<<dim3(ROWS / 64, CC / 64), 128, 0, stream>>>(ob, wpb, b_proj, out);
}
